// HadamardQuantLinear_90443421319612
// MI455X (gfx1250) — 8-bit matrix operand order repaired and checked by reading, compiled, not run on hardware
//
#include <hip/hip_runtime.h>
#include <stdint.h>

typedef __attribute__((ext_vector_type(8))) int v8i;

#define IN_F   2048
#define OUT_F  2048
#define NROWS  16384


__global__ void init_kernel(unsigned* mb) {
    mb[0] = 0u;
    mb[1] = 0u;
}

__global__ void had_absmax_kernel(const float* __restrict__ src,
                                  unsigned* __restrict__ maxbits) {
    __shared__ float sh[256];
    __shared__ float red[256];
    int t = threadIdx.x;
    long long gbase = (long long)blockIdx.x * 256;
    sh[t] = src[gbase + t];
    __syncthreads();
    int b = t >> 6;
    int j = t & 63;
    const float* p = &sh[b * 64];
    float acc = 0.0f;
#pragma unroll
    for (int i = 0; i < 64; ++i) {
        float v = p[i];
        acc += (__popc(i & j) & 1) ? -v : v;
    }
    acc *= 0.125f;
    red[t] = fabsf(acc);
    __syncthreads();
    for (int s = 128; s > 0; s >>= 1) {
        if (t < s) red[t] = fmaxf(red[t], red[t + s]);
        __syncthreads();
    }
    if (t == 0) atomicMax(maxbits, __float_as_uint(red[0]));
}

__global__ void finalize_scales_kernel(unsigned* mb) {
    const float tiny = 1.17549435082228750797e-38f;
    float mx = __uint_as_float(mb[0]);
    float mw = __uint_as_float(mb[1]);
    ((float*)mb)[2] = fmaxf(mx / 127.0f, tiny);
    ((float*)mb)[3] = fmaxf(mw / 127.0f, tiny);
}

__global__ void had_quant_kernel(const float* __restrict__ src,
                                 const float* __restrict__ scale,
                                 int8_t* __restrict__ dst) {
    __shared__ float sh[256];
    int t = threadIdx.x;
    long long gbase = (long long)blockIdx.x * 256;
    sh[t] = src[gbase + t];
    __syncthreads();
    int b = t >> 6;
    int j = t & 63;
    const float* p = &sh[b * 64];
    float acc = 0.0f;
#pragma unroll
    for (int i = 0; i < 64; ++i) {
        float v = p[i];
        acc += (__popc(i & j) & 1) ? -v : v;
    }
    acc *= 0.125f;
    float q = rintf(acc / *scale);
    q = fminf(fmaxf(q, -127.0f), 127.0f);
    dst[gbase + t] = (int8_t)q;
}

#define LDS_STRIDE_B 80
#define LDS_STRIDE_W 20

#define ASYNC_B128(ldsoff, gaddr)                                   \
    asm volatile("global_load_async_to_lds_b128 %0, %1, off"        \
                 :: "v"(ldsoff), "v"(gaddr) : "memory")

__global__ __launch_bounds__(256)
void gemm_i8_wmma_kernel(const int8_t* __restrict__ Aq,
                         const int8_t* __restrict__ Bq,
                         const float*  __restrict__ scales,
                         const float*  __restrict__ bias,
                         float*        __restrict__ out) {
    __shared__ int8_t lA[2][128 * LDS_STRIDE_B];
    __shared__ int8_t lB[2][128 * LDS_STRIDE_B];

    const int t    = threadIdx.x;
    const int lane = t & 31;
    const int w    = t >> 5;
    const int wr   = w & 3;
    const int wc   = w >> 2;
    const int l    = lane & 15;
    const int hi   = lane >> 4;

    const long long rowBase = (long long)blockIdx.y * 128;
    const long long colBase = (long long)blockIdx.x * 128;

    const int ra0 = t >> 2,         ka0 = (t & 3) * 16;
    const int ra1 = (t + 256) >> 2, ka1 = ((t + 256) & 3) * 16;

    const int8_t* gA0 = Aq + (rowBase + ra0) * IN_F + ka0;
    const int8_t* gA1 = Aq + (rowBase + ra1) * IN_F + ka1;
    const int8_t* gB0 = Bq + (colBase + ra0) * IN_F + ka0;
    const int8_t* gB1 = Bq + (colBase + ra1) * IN_F + ka1;

    unsigned la0[2], la1[2], lb0[2], lb1[2];
#pragma unroll
    for (int p = 0; p < 2; ++p) {
        la0[p] = (unsigned)(size_t)&lA[p][ra0 * LDS_STRIDE_B + ka0];
        la1[p] = (unsigned)(size_t)&lA[p][ra1 * LDS_STRIDE_B + ka1];
        lb0[p] = (unsigned)(size_t)&lB[p][ra0 * LDS_STRIDE_B + ka0];
        lb1[p] = (unsigned)(size_t)&lB[p][ra1 * LDS_STRIDE_B + ka1];
    }

    auto issue_tile = [&](int kOff, int p) {
        ASYNC_B128(la0[p], (unsigned long long)(size_t)(gA0 + kOff));
        ASYNC_B128(la1[p], (unsigned long long)(size_t)(gA1 + kOff));
        ASYNC_B128(lb0[p], (unsigned long long)(size_t)(gB0 + kOff));
        ASYNC_B128(lb1[p], (unsigned long long)(size_t)(gB1 + kOff));
    };

    v8i acc[2][4] = {};

    issue_tile(0, 0);

    for (int k0 = 0; k0 < IN_F; k0 += 64) {
        const int p = (k0 >> 6) & 1;

        asm volatile("s_wait_asynccnt 0x0" ::: "memory");
        __syncthreads();

        if (k0 + 64 < IN_F) issue_tile(k0 + 64, p ^ 1);

        const unsigned* lAw = (const unsigned*)lA[p];
        const unsigned* lBw = (const unsigned*)lB[p];

        v8i af[2], bf[4];
#pragma unroll
        for (int i = 0; i < 2; ++i) {
            int rowM = wr * 32 + i * 16 + l;
#pragma unroll
            for (int v = 0; v < 8; ++v) {
                int wi = (v >> 1) * 4 + hi * 2 + (v & 1);
                af[i][v] = (int)lAw[rowM * LDS_STRIDE_W + wi];
            }
        }
#pragma unroll
        for (int j = 0; j < 4; ++j) {
            int rowO = wc * 64 + j * 16 + l;
#pragma unroll
            for (int v = 0; v < 8; ++v) {
                int wi = (v >> 1) * 4 + hi * 2 + (v & 1);
                bf[j][v] = (int)lBw[rowO * LDS_STRIDE_W + wi];
            }
        }

#pragma unroll
        for (int i = 0; i < 2; ++i)
#pragma unroll
            for (int j = 0; j < 4; ++j)
                acc[i][j] = __builtin_amdgcn_wmma_i32_16x16x64_iu8(
                     true, af[i],  true, bf[j],
                    acc[i][j],  false,  false);
    }

    const float s = scales[2] * scales[3];
#pragma unroll
    for (int i = 0; i < 2; ++i) {
#pragma unroll
        for (int j = 0; j < 4; ++j) {
            long long col = colBase + wc * 64 + j * 16 + l;
            float bv = bias[col];
#pragma unroll
            for (int r = 0; r < 8; ++r) {
                long long row = rowBase + wr * 32 + i * 16 + hi * 8 + r;
                out[row * OUT_F + col] = (float)acc[i][j][r] * s + bv;
            }
        }
    }
}

extern "C" void kernel_launch(void* const* d_in, const int* in_sizes, int n_in,
                              void* d_out, int out_size, void* d_ws, size_t ws_size,
                              hipStream_t stream) {
    const float* x      = (const float*)d_in[0];
    const float* weight = (const float*)d_in[1];
    const float* bias   = (const float*)d_in[2];

    int8_t*   xq  = (int8_t*)d_ws;
    int8_t*   wq  = xq + (size_t)NROWS * IN_F;
    unsigned* mb  = (unsigned*)(wq + (size_t)OUT_F * IN_F);
    float*    scl = (float*)mb;
    float*    out = (float*)d_out;

    init_kernel<<<1, 1, 0, stream>>>(mb);

    had_absmax_kernel<<<(NROWS * IN_F) / 256, 256, 0, stream>>>(x, mb + 0);
    had_absmax_kernel<<<(OUT_F * IN_F) / 256, 256, 0, stream>>>(weight, mb + 1);

    finalize_scales_kernel<<<1, 1, 0, stream>>>(mb);

    had_quant_kernel<<<(NROWS * IN_F) / 256, 256, 0, stream>>>(x, scl + 2, xq);
    had_quant_kernel<<<(OUT_F * IN_F) / 256, 256, 0, stream>>>(weight, scl + 3, wq);

    dim3 grid(OUT_F / 128, NROWS / 128);
    gemm_i8_wmma_kernel<<<grid, 256, 0, stream>>>(xq, wq, scl, bias, out);
}
